// PhysicsEngine_45664092291394
// MI455X (gfx1250) — matrix operand order repaired and checked by reading, not compiled, not run on hardware
//
#include <hip/hip_runtime.h>
#include <hip/hip_bf16.h>

#define NPTS 16384
#define KMAX 64
#define RAD2 (0.025f * 0.025f)

typedef __bf16 bf16_t;
typedef bf16_t v16bf __attribute__((ext_vector_type(16)));
typedef bf16_t v8bf  __attribute__((ext_vector_type(8)));
typedef float  v8f   __attribute__((ext_vector_type(8)));

// ---------------------------------------------------------------------------
// Branch-free erf (Abramowitz-Stegun 7.1.26, |err| < 1.5e-7):
//   one v_rcp_f32 + one v_exp_f32 + 6 FMAs, no EXEC-mask divergence.
// Replaces the OCML erff() which expands to a v_cmpx/saveexec branchy ladder
// (268M calls in the WMMA epilogue).
// ---------------------------------------------------------------------------
__device__ __forceinline__ float erf_fast(float x) {
    float ax = __builtin_fabsf(x);
    float t  = __builtin_amdgcn_rcpf(__builtin_fmaf(0.3275911f, ax, 1.0f));
    float y  = __builtin_fmaf(t, 1.061405429f, -1.453152027f);
    y = __builtin_fmaf(t, y,  1.421413741f);
    y = __builtin_fmaf(t, y, -0.284496736f);
    y = __builtin_fmaf(t, y,  0.254829592f);
    y = y * t;
    float e  = __expf(-ax * ax);                 // v_exp_f32, branch-free
    float r  = __builtin_fmaf(-y, e, 1.0f);
    return __builtin_copysignf(r, x);
}

__device__ __forceinline__ float gelu_fast(float x) {
    return 0.5f * x * (1.0f + erf_fast(x * 0.70710678118654752f));
}

// ---------------------------------------------------------------------------
// Kernel 0: W2 [128,256] f32 (k-major) -> W2T [256,128] bf16 (n-major) so that
// a WMMA B-fragment (fixed column n, two runs of 8 consecutive k) is two contiguous 16B loads.
// ---------------------------------------------------------------------------
__global__ void k_convert_w2(const float* __restrict__ W2, bf16_t* __restrict__ W2T) {
    int e = blockIdx.x * blockDim.x + threadIdx.x;   // 0..32767
    int k = e >> 8;                                  // 0..127
    int n = e & 255;                                 // 0..255
    W2T[n * 128 + k] = (bf16_t)W2[e];
}

// ---------------------------------------------------------------------------
// Kernel 1: brute-force radius neighbor list. rom_ic (192KB) stays in L2; all
// lanes step through the same j, so loads are broadcast/cache friendly.
// ---------------------------------------------------------------------------
__global__ void k_neighbors(const float* __restrict__ y, const float* __restrict__ x,
                            int* __restrict__ idx, int* __restrict__ cnt) {
    int q = blockIdx.x * blockDim.x + threadIdx.x;
    if (q >= NPTS) return;
    float qx = x[q * 3 + 0], qy = x[q * 3 + 1], qz = x[q * 3 + 2];
    int c = 0;
    for (int j = 0; j < NPTS; ++j) {
        float dx = y[j * 3 + 0] - qx;
        float dy = y[j * 3 + 1] - qy;
        float dz = y[j * 3 + 2] - qz;
        float d2 = dx * dx + dy * dy + dz * dz;
        if (d2 <= RAD2) {
            idx[q * KMAX + c] = j;
            if (++c == KMAX) break;
        }
    }
    cnt[q] = c;
    for (int s = c; s < KMAX; ++s) idx[q * KMAX + s] = -1;
}

// ---------------------------------------------------------------------------
// Kernel 2: fused edge-MLP + aggregation. One wave per query; 4 tiles of
// M=16 edge slots. Layer 2 (128->256) runs on v_wmma_f32_16x16x32_bf16.
// ---------------------------------------------------------------------------
__launch_bounds__(256)
__global__ void k_gno_mlp(const float* __restrict__ y, const float* __restrict__ x,
                          const float* __restrict__ f,
                          const float* __restrict__ W1, const float* __restrict__ b1,
                          const bf16_t* __restrict__ W2T, const float* __restrict__ b2,
                          const float* __restrict__ W3, const float* __restrict__ b3,
                          const int* __restrict__ idx, const int* __restrict__ cnt,
                          float* __restrict__ out) {
    // per-wave LDS regions (8 waves / block)
    __shared__ __align__(16) bf16_t h1[8][16][136];  // 16 rows x 128 (+8 pad) bf16
    __shared__ float rel_s[8][16][3];
    __shared__ float kout[8][16][3];
    __shared__ float qsum[8][4];
    __shared__ int   sidx[8][16];

    const int w    = threadIdx.x >> 5;
    const int lane = threadIdx.x & 31;
    const int q    = blockIdx.x * 8 + w;
    const int m    = lane & 15;   // row (A) / col (B,C) owned by this lane
    const int hi   = lane >> 4;   // lane-half selects K/M sub-blocks

    __builtin_prefetch(W2T, 0, 3);

    if (lane < 4) qsum[w][lane] = 0.0f;

    const float qx = x[q * 3 + 0], qy = x[q * 3 + 1], qz = x[q * 3 + 2];

    for (int t = 0; t < 4; ++t) {
        // ---- gather 16 edge slots: rel = y[idx] - x ----
        if (lane < 16) {
            int id = idx[q * KMAX + t * 16 + lane];
            sidx[w][lane] = id;
            float rx = 0.f, ry = 0.f, rz = 0.f;
            if (id >= 0) {
                rx = y[id * 3 + 0] - qx;
                ry = y[id * 3 + 1] - qy;
                rz = y[id * 3 + 2] - qz;
            }
            rel_s[w][lane][0] = rx; rel_s[w][lane][1] = ry; rel_s[w][lane][2] = rz;
        }
        for (int v = lane; v < 48; v += 32) kout[w][v / 3][v % 3] = 0.0f;
        __syncthreads();

        // ---- layer 1: [16,3]@[3,128] + GELU, bf16 A-tile into LDS ----
        for (int v = lane; v < 2048; v += 32) {
            int s = v >> 7, j = v & 127;
            float a = rel_s[w][s][0] * W1[j]
                    + rel_s[w][s][1] * W1[128 + j]
                    + rel_s[w][s][2] * W1[256 + j] + b1[j];
            h1[w][s][j] = (bf16_t)gelu_fast(a);
        }
        __syncthreads();

        // ---- build 4 A fragments (K=0..127 in chunks of 32), ISA wave32 layout:
        // lanes 0-15: halves = K[kc*32 + 0..7], K[kc*32 + 16..23] of row m
        // lanes 16-31: halves = K[kc*32 + 8..15], K[kc*32 + 24..31] of row m
        v16bf afrag[4];
        #pragma unroll
        for (int kc = 0; kc < 4; ++kc) {
            v8bf lo = *reinterpret_cast<const v8bf*>(&h1[w][m][kc * 32 + hi * 8]);
            v8bf up = *reinterpret_cast<const v8bf*>(&h1[w][m][kc * 32 + 16 + hi * 8]);
            #pragma unroll
            for (int i = 0; i < 8; ++i) { afrag[kc][i] = lo[i]; afrag[kc][8 + i] = up[i]; }
        }

        float acc3[8][3];
        #pragma unroll
        for (int r = 0; r < 8; ++r)
            for (int c = 0; c < 3; ++c) acc3[r][c] = 0.0f;

        // ---- layer 2: 16 N-tiles x 4 K-steps of WMMA; epilogue folds b2,
        // GELU and the 256->3 layer-3 dot product (lane owns column ncol) ----
        for (int nt = 0; nt < 16; ++nt) {
            const int ncol = nt * 16 + m;
            const bf16_t* brow = W2T + (size_t)ncol * 128 + hi * 8;
            v8f cacc = {};
            #pragma unroll
            for (int kc = 0; kc < 4; ++kc) {
                const v8bf* p = reinterpret_cast<const v8bf*>(brow + kc * 32);
                v8bf lo = p[0], up = p[2];
                v16bf bfrag;
                #pragma unroll
                for (int i = 0; i < 8; ++i) { bfrag[i] = lo[i]; bfrag[8 + i] = up[i]; }
                cacc = __builtin_amdgcn_wmma_f32_16x16x32_bf16(
                    false, afrag[kc], false, bfrag, (short)0, cacc, false, false);
            }
            const float bias = b2[ncol];
            const float w30 = W3[ncol * 3 + 0];
            const float w31 = W3[ncol * 3 + 1];
            const float w32 = W3[ncol * 3 + 2];
            #pragma unroll
            for (int r = 0; r < 8; ++r) {  // C layout: VGPR r = row hi*8+r, col ncol
                float g = gelu_fast(cacc[r] + bias);
                acc3[r][0] = __builtin_fmaf(g, w30, acc3[r][0]);
                acc3[r][1] = __builtin_fmaf(g, w31, acc3[r][1]);
                acc3[r][2] = __builtin_fmaf(g, w32, acc3[r][2]);
            }
        }

        // ---- reduce layer-3 partials across the 16 columns each lane-half saw ----
        #pragma unroll
        for (int r = 0; r < 8; ++r)
            for (int c = 0; c < 3; ++c)
                atomicAdd(&kout[w][hi * 8 + r][c], acc3[r][c]);
        __syncthreads();

        // ---- vals = (k + b3) * f_y, masked accumulate into the query sum ----
        if (lane < 16) {
            int id = sidx[w][lane];
            if (id >= 0) {
                for (int c = 0; c < 3; ++c) {
                    float v = (kout[w][lane][c] + b3[c]) * f[id * 3 + c];
                    atomicAdd(&qsum[w][c], v);
                }
            }
        }
        __syncthreads();
    }

    if (lane == 0) {
        float cf = (float)cnt[q];
        if (cf < 1.0f) cf = 1.0f;
        for (int c = 0; c < 3; ++c) out[q * 3 + c] = qsum[w][c] / cf;
    }
}

// ---------------------------------------------------------------------------
extern "C" void kernel_launch(void* const* d_in, const int* in_sizes, int n_in,
                              void* d_out, int out_size, void* d_ws, size_t ws_size,
                              hipStream_t stream) {
    (void)in_sizes; (void)n_in; (void)out_size; (void)ws_size;
    const float* rom_ic = (const float*)d_in[0];
    const float* fom_ic = (const float*)d_in[1];
    const float* rom_f  = (const float*)d_in[2];
    const float* W1 = (const float*)d_in[3];
    const float* b1 = (const float*)d_in[4];
    const float* W2 = (const float*)d_in[5];
    const float* b2 = (const float*)d_in[6];
    const float* W3 = (const float*)d_in[7];
    const float* b3 = (const float*)d_in[8];
    float* out = (float*)d_out;

    char* ws = (char*)d_ws;
    bf16_t* W2T = (bf16_t*)ws;               // 256*128*2      = 65536 B
    int* cnt    = (int*)(ws + 65536);        // 16384*4        = 65536 B
    int* idx    = (int*)(ws + 131072);       // 16384*64*4     = 4 MiB

    k_convert_w2<<<128, 256, 0, stream>>>(W2, W2T);
    k_neighbors<<<NPTS / 256, 256, 0, stream>>>(rom_ic, fom_ic, idx, cnt);
    k_gno_mlp<<<NPTS / 8, 256, 0, stream>>>(rom_ic, fom_ic, rom_f,
                                            W1, b1, W2T, b2, W3, b3,
                                            idx, cnt, out);
}
